// SFAttentionNetwork_30966714204894
// MI455X (gfx1250) — matrix operand order repaired and checked by reading, not compiled, not run on hardware
//
#include <hip/hip_runtime.h>
#include <math.h>

// ---------------- problem constants ----------------
#define B_SZ   16384
#define NB     8
#define NA     16
#define DD     128
#define IN_DIM 2304        // DD + NA*DD + DD
#define QK_DIM 384
#define V_DIM  128
#define KSTEPS 72          // IN_DIM / 32

// ---------------- workspace byte offsets ----------------
#define WS_INVNORM 0                        // 131072 f32
#define WS_WKT     524288                   // 384*2304 bf16
#define WS_WQT     (WS_WKT + 1769472)       // 384*2304 bf16
#define WS_WVT     (WS_WQT + 1769472)       // 128*128 bf16
#define WS_QUERY   (WS_WVT + 32768)         // 16384*384 f32

// ---------------- d_out float offsets (reference tuple order) ----------------
#define OUT_Q1      0                       // (B,16)
#define OUT_ATT     262144                  // (B,1,16,128)
#define OUT_LOGITS  33816576                // (B,1,8)
#define OUT_WEIGHTS 33947648                // (B,1,8)
#define OUT_KEYS    34078720                // (B,8,384)
#define OUT_VALS    84410368                // (B,8,16,128)
#define OUT_SIM     352845824               // (7,)

typedef __attribute__((ext_vector_type(16))) __bf16 v16bf;
typedef __attribute__((ext_vector_type(8)))  float  v8f;

union Frag16 { v16bf v; uint4 q[2]; };

__device__ __forceinline__ unsigned short f32_to_bf16(float f) {
  unsigned int u = __float_as_uint(f);
  u = (u + 0x7FFFu + ((u >> 16) & 1u)) >> 16;   // round-to-nearest-even
  return (unsigned short)u;
}

// ---------------------------------------------------------------------------
// K0: weight transpose + f32->bf16 (N-major so B-fragments are contiguous),
//     zero sim accumulators.
// ---------------------------------------------------------------------------
__global__ void k_prep(const float* __restrict__ Wq, const float* __restrict__ Wk,
                       const float* __restrict__ Wv,
                       unsigned short* __restrict__ WqT, unsigned short* __restrict__ WkT,
                       unsigned short* __restrict__ WvT, float* __restrict__ simg) {
  const int stride = gridDim.x * blockDim.x;
  const int g = blockIdx.x * blockDim.x + threadIdx.x;
  for (int idx = g; idx < IN_DIM * QK_DIM; idx += stride) {
    int n = idx / IN_DIM, k = idx % IN_DIM;           // WT[n][k] = W[k][n]
    WkT[idx] = f32_to_bf16(Wk[(size_t)k * QK_DIM + n]);
    WqT[idx] = f32_to_bf16(Wq[(size_t)k * QK_DIM + n]);
  }
  for (int idx = g; idx < DD * V_DIM; idx += stride) {
    int n = idx / DD, k = idx % DD;
    WvT[idx] = f32_to_bf16(Wv[(size_t)k * V_DIM + n]);
  }
  if (g < 7) simg[g] = 0.0f;
}

// ---------------------------------------------------------------------------
// K1: inv_norm[b*NB+nb] = 1/||task[b,nb,:]||  (wave per row, wave32 shuffle)
// ---------------------------------------------------------------------------
__global__ void k_invnorm(const float* __restrict__ task, float* __restrict__ invn) {
  const int row  = blockIdx.x * 8 + (threadIdx.x >> 5);
  const int lane = threadIdx.x & 31;
  const float* t = task + (size_t)row * DD;
  float s = 0.0f;
  for (int i = lane; i < DD; i += 32) { float v = t[i]; s += v * v; }
  for (int off = 16; off > 0; off >>= 1) s += __shfl_xor(s, off, 32);
  if (lane == 0) invn[row] = 1.0f / sqrtf(s);
}

// ---------------------------------------------------------------------------
// K2/K3: GEMM over k_in = [basis | sf | task_n]  (built on the fly in LDS).
// MODE 0: keys_masked = (k_in @ Wk + bk) * mask_rep, 16 rows = 2 batches x 8nb,
//         with the rbf-sim reduction fused into the f32 A-staging buffer.
// MODE 1: query = k_in[:,0,:] @ Wq, 16 rows = 16 batches' nb==0 row.
// ---------------------------------------------------------------------------
template <int MODE>
__global__ void k_gemm(const float* __restrict__ basis,
                       const float* __restrict__ sf,
                       const float* __restrict__ task,
                       const float* __restrict__ mask,
                       const float* __restrict__ bk,
                       const float* __restrict__ invn,
                       const unsigned short* __restrict__ wT,
                       float* __restrict__ outp,
                       float* __restrict__ simg) {
  __shared__ float          Asf[16][32];          // f32 staging (== bst slice)
  __shared__ unsigned short Abf[16][32];          // bf16 A tile
  __shared__ unsigned short Bs[QK_DIM][32];       // bf16 B tile (n-major)
  __shared__ float          simacc[14];

  const int tid  = threadIdx.x;
  const int lane = tid & 31;
  const int wave = tid >> 5;
  const int nlo  = lane & 15;
  const int kh   = lane >> 4;

  if (MODE == 0 && tid < 14) simacc[tid] = 0.0f;

  v8f acc[3] = {};
  float sp0 = 0.0f, sp1 = 0.0f;

  for (int ks = 0; ks < KSTEPS; ++ks) {
    const int k0 = ks * 32;
    // ---- stage A: gather k_in slice, keep f32 (for sim) + bf16 (for WMMA) ----
#pragma unroll
    for (int ii = 0; ii < 2; ++ii) {
      const int idx = tid + ii * 256;
      const int r = idx >> 5, c = idx & 31;
      const int cg = k0 + c;
      int b, nb;
      if (MODE == 0) { b = blockIdx.x * 2 + (r >> 3); nb = r & 7; }
      else           { b = blockIdx.x * 16 + r;       nb = 0;     }
      const int row = b * NB + nb;
      float f;
      if (cg < DD)                 f = basis[(size_t)row * DD + cg];
      else if (cg < DD + NA * DD)  f = sf[(size_t)b * (NB * NA * DD) +
                                          (size_t)nb * (NA * DD) + (cg - DD)];
      else                         f = task[(size_t)row * DD + (cg - DD - NA * DD)] * invn[row];
      Asf[r][c] = f;
      Abf[r][c] = f32_to_bf16(f);
    }
    // ---- stage B: 32-K slice of all 384 N rows (contiguous 64B per row) ----
    for (int n = tid; n < QK_DIM; n += 256) {
      const uint4* src = (const uint4*)(wT + (size_t)n * IN_DIM + k0);
      uint4* dst = (uint4*)&Bs[n][0];
      dst[0] = src[0]; dst[1] = src[1]; dst[2] = src[2]; dst[3] = src[3];
      __builtin_prefetch((const void*)(src + 4), 0, 0);   // next K slice
    }
    __syncthreads();

    // ---- fused rbf-sim partial sums (reads the f32 bst slice) ----
    if (MODE == 0) {
      { const int p = tid >> 5, c = tid & 31;
        const int r0 = (p / 7) * 8 + (p % 7);
        const float d = Asf[r0][c] - Asf[r0 + 1][c];
        sp0 += d * d; }
      if (tid < 192) {
        const int idx = tid + 256;
        const int p = idx >> 5, c = idx & 31;
        const int r0 = (p / 7) * 8 + (p % 7);
        const float d = Asf[r0][c] - Asf[r0 + 1][c];
        sp1 += d * d;
      }
    }

    // ---- WMMA: 8 waves x 3 N-tiles, one shared A fragment per wave ----
    Frag16 af;
    af.q[0] = *(const uint4*)&Abf[nlo][kh * 8];          // K kh*8 .. +7
    af.q[1] = *(const uint4*)&Abf[nlo][16 + kh * 8];     // K 16+kh*8 .. +7
#pragma unroll
    for (int t = 0; t < 3; ++t) {
      const int n = (wave + 8 * t) * 16 + nlo;
      Frag16 bf;
      bf.q[0] = *(const uint4*)&Bs[n][kh * 8];           // K kh*8 .. +7
      bf.q[1] = *(const uint4*)&Bs[n][16 + kh * 8];      // K 16+kh*8 .. +7
      acc[t] = __builtin_amdgcn_wmma_f32_16x16x32_bf16(
          false, af.v, false, bf.v, (short)0, acc[t], false, false);
    }
    __syncthreads();
  }

  // ---- epilogue ----
#pragma unroll
  for (int t = 0; t < 3; ++t) {
    const int n = (wave + 8 * t) * 16 + nlo;
    const float bias = (MODE == 0) ? bk[n] : 0.0f;
#pragma unroll
    for (int v = 0; v < 8; ++v) {
      const int r = v + 8 * kh;                          // C/D row layout
      if (MODE == 0) {
        const int b = blockIdx.x * 2 + (r >> 3);
        const int nb = r & 7;
        const float m = mask[(size_t)(b * NB + nb) * DD + n / 3];  // repeat(mask,3)
        outp[(size_t)(b * NB + nb) * QK_DIM + n] = (acc[t][v] + bias) * m;
      } else {
        const int b = blockIdx.x * 16 + r;
        outp[(size_t)b * QK_DIM + n] = acc[t][v];
      }
    }
  }

  if (MODE == 0) {
    atomicAdd(&simacc[tid >> 5], sp0);
    if (tid < 192) atomicAdd(&simacc[(tid >> 5) + 8], sp1);
    __syncthreads();
    if (tid < 14)
      atomicAdd(&simg[tid % 7], expf(-0.5f * simacc[tid]) * (1.0f / (float)B_SZ));
  }
}

// ---------------------------------------------------------------------------
// K4: per-batch fused values GEMM (WMMA bf16) + logits + softmax + attended + q1
// ---------------------------------------------------------------------------
__global__ void k_attn(const float* __restrict__ sf,
                       const float* __restrict__ task,
                       const float* __restrict__ bv,
                       const unsigned short* __restrict__ wvT,
                       const float* __restrict__ query,
                       const float* __restrict__ keysm,
                       float* __restrict__ out_q1,
                       float* __restrict__ out_att,
                       float* __restrict__ out_logit,
                       float* __restrict__ out_w,
                       float* __restrict__ out_val) {
  extern __shared__ char smem[];
  unsigned short* Abf = (unsigned short*)smem;             // [128][128] bf16 sf
  unsigned short* Bsv = (unsigned short*)(smem + 32768);   // [128][128] bf16 WvT
  float* Vs  = (float*)(smem + 65536);                     // [8][16][128] f32 values
  float* Att = (float*)(smem + 131072);                    // [16][128]
  float* tk  = (float*)(smem + 139264);                    // [128] task_first
  float* lg  = (float*)(smem + 139776);                    // [8]  logits

  const int tid  = threadIdx.x;
  const int lane = tid & 31;
  const int wave = tid >> 5;
  const int nlo  = lane & 15;
  const int kh   = lane >> 4;
  const int b    = blockIdx.x;

  const float* sfb = sf + (size_t)b * (NB * NA * DD);
  for (int idx = tid; idx < NB * NA * DD; idx += 256) Abf[idx] = f32_to_bf16(sfb[idx]);
  for (int idx = tid; idx < DD * V_DIM; idx += 256)   Bsv[idx] = wvT[idx];
  if (tid < DD) tk[tid] = task[(size_t)b * (NB * DD) + tid];   // task[b,0,:]
  __syncthreads();

  // values: wave w owns a == w (M rows 16w..16w+15), 8 N-tiles x 4 K-steps
  const int rbase = wave * 16;
#pragma unroll
  for (int nt = 0; nt < 8; ++nt) {
    v8f acc = {};
    const int n = nt * 16 + nlo;
#pragma unroll
    for (int kt = 0; kt < 4; ++kt) {
      const int k0 = kt * 32;
      Frag16 af, bf;
      af.q[0] = *(const uint4*)&Abf[(rbase + nlo) * DD + k0 + kh * 8];
      af.q[1] = *(const uint4*)&Abf[(rbase + nlo) * DD + k0 + 16 + kh * 8];
      bf.q[0] = *(const uint4*)&Bsv[n * DD + k0 + kh * 8];
      bf.q[1] = *(const uint4*)&Bsv[n * DD + k0 + 16 + kh * 8];
      acc = __builtin_amdgcn_wmma_f32_16x16x32_bf16(
          false, af.v, false, bf.v, (short)0, acc, false, false);
    }
    const float bias = bv[n];
#pragma unroll
    for (int v = 0; v < 8; ++v) {
      const int na = v + 8 * kh;
      const float val = acc[v] + bias;
      Vs[(wave * 16 + na) * DD + n] = val;
      out_val[(size_t)b * (NB * NA * DD) + (size_t)(wave * 16 + na) * DD + n] = val;
    }
  }
  __syncthreads();

  // logits: wave a computes query . keys_masked[a] / sqrt(128)
  {
    const float* q = query + (size_t)b * QK_DIM;
    const float* k = keysm + (size_t)(b * NB + wave) * QK_DIM;
    float part = 0.0f;
    for (int n = lane; n < QK_DIM; n += 32) part += q[n] * k[n];
    for (int off = 16; off > 0; off >>= 1) part += __shfl_xor(part, off, 32);
    if (lane == 0) {
      float l = part * 0.088388347648318447f;   // 1/sqrt(128)
      if (l == 0.0f) l = -1.0e9f;               // where(logits==0, -1e9)
      lg[wave] = l;
    }
  }
  __syncthreads();

  // softmax over 8 (redundant per thread; tiny)
  float w8[8];
  {
    float m = lg[0];
#pragma unroll
    for (int a = 1; a < 8; ++a) m = fmaxf(m, lg[a]);
    float s = 0.0f;
#pragma unroll
    for (int a = 0; a < 8; ++a) { w8[a] = expf(lg[a] - m); s += w8[a]; }
    const float inv = 1.0f / s;
#pragma unroll
    for (int a = 0; a < 8; ++a) w8[a] *= inv;
  }
  if (tid < 8) {
    out_logit[(size_t)b * NB + tid] = lg[tid];
    out_w[(size_t)b * NB + tid]     = w8[tid];
  }

  // attended = sum_a w[a] * values[a]  (16x128)
  for (int idx = tid; idx < NA * DD; idx += 256) {
    float s = 0.0f;
#pragma unroll
    for (int a = 0; a < 8; ++a) s += w8[a] * Vs[a * (NA * DD) + idx];
    Att[idx] = s;
    out_att[(size_t)b * (NA * DD) + idx] = s;
  }
  __syncthreads();

  // q_1[j] = sum_i task_first[i] * attended[j][i]
  if (tid < NA) {
    float s = 0.0f;
    for (int i = 0; i < DD; ++i) s += tk[i] * Att[tid * DD + i];
    out_q1[(size_t)b * NA + tid] = s;
  }
}

// ---------------------------------------------------------------------------
extern "C" void kernel_launch(void* const* d_in, const int* in_sizes, int n_in,
                              void* d_out, int out_size, void* d_ws, size_t ws_size,
                              hipStream_t stream) {
  const float* basis = (const float*)d_in[0];
  const float* sf    = (const float*)d_in[1];
  const float* task  = (const float*)d_in[2];
  const float* mask  = (const float*)d_in[3];
  const float* Wq    = (const float*)d_in[4];
  const float* Wk    = (const float*)d_in[5];
  const float* bk    = (const float*)d_in[6];
  const float* Wv    = (const float*)d_in[7];
  const float* bv    = (const float*)d_in[8];
  float* out = (float*)d_out;
  char*  ws  = (char*)d_ws;

  float*          invn  = (float*)(ws + WS_INVNORM);
  unsigned short* WkT   = (unsigned short*)(ws + WS_WKT);
  unsigned short* WqT   = (unsigned short*)(ws + WS_WQT);
  unsigned short* WvT   = (unsigned short*)(ws + WS_WVT);
  float*          query = (float*)(ws + WS_QUERY);

  k_prep<<<512, 256, 0, stream>>>(Wq, Wk, Wv, WqT, WkT, WvT, out + OUT_SIM);
  k_invnorm<<<(B_SZ * NB) / 8, 256, 0, stream>>>(task, invn);
  k_gemm<0><<<B_SZ / 2, 256, 0, stream>>>(basis, sf, task, mask, bk, invn, WkT,
                                          out + OUT_KEYS, out + OUT_SIM);
  k_gemm<1><<<B_SZ / 16, 256, 0, stream>>>(basis, sf, task, nullptr, nullptr, invn, WqT,
                                           query, nullptr);
  (void)hipFuncSetAttribute((const void*)k_attn,
                            hipFuncAttributeMaxDynamicSharedMemorySize, 139840);
  k_attn<<<B_SZ, 256, 139840, stream>>>(sf, task, bv, WvT, query, out + OUT_KEYS,
                                        out + OUT_Q1, out + OUT_ATT, out + OUT_LOGITS,
                                        out + OUT_WEIGHTS, out + OUT_VALS);
}
